// TorchLSTMRegressor_83416854823505
// MI455X (gfx1250) — matrix operand order repaired and checked by reading, not compiled, not run on hardware
//
#include <hip/hip_runtime.h>

typedef __attribute__((ext_vector_type(16))) _Float16     v16h;
typedef __attribute__((ext_vector_type(8)))  float        v8f;
typedef __attribute__((ext_vector_type(8)))  unsigned int v8u;

#define HID   32
#define T_LEN 512

__device__ __forceinline__ float fast_sigmoid(float x) {
    // 1/(1+exp(-x)) via hardware exp2/rcp
    return __builtin_amdgcn_rcpf(1.0f + __builtin_amdgcn_exp2f(-1.4426950408889634f * x));
}
__device__ __forceinline__ float fast_tanh(float x) {
    return 2.0f * __builtin_amdgcn_rcpf(1.0f + __builtin_amdgcn_exp2f(-2.885390081777927f * x)) - 1.0f;
}

__device__ __forceinline__ unsigned int pack_h2(_Float16 a, _Float16 b) {
    unsigned short ua = __builtin_bit_cast(unsigned short, a);
    unsigned short ub = __builtin_bit_cast(unsigned short, b);
    return (unsigned int)ua | ((unsigned int)ub << 16);
}

// Overwrite only dword0 of a v16h operand (upper 7 dwords stay shared zeros).
__device__ __forceinline__ v16h set_dword0(v16h v, unsigned int w) {
    v8u u = __builtin_bit_cast(v8u, v);
    u[0] = w;
    return __builtin_bit_cast(v16h, u);
}

// One wave (32 lanes) per 16-batch tile. Transposed LSTM recurrence:
//   gates^T[128 x 16] = Whh_perm[128 x 32] @ h^T[32 x 16]  (+ rank-1 x/bias WMMA)
// Row permutation gate_row = 32*g + 16*hi + 2*r + p makes the WMMA D-layout of
// h identical to the B-layout required next step -> no shuffles, no LDS in loop.
__global__ __launch_bounds__(32)
void lstm_wmma_kernel(const float* __restrict__ x,
                      const float* __restrict__ W_ih,
                      const float* __restrict__ W_hh,
                      const float* __restrict__ b_ih,
                      const float* __restrict__ b_hh,
                      const float* __restrict__ fc1_w,
                      const float* __restrict__ fc1_b,
                      const float* __restrict__ fc2_w,
                      const float* __restrict__ fc2_b,
                      float* __restrict__ out,
                      int Tt)
{
    const int lane = threadIdx.x;          // wave32
    const int m    = lane & 15;            // A-row / batch column index
    const int hi   = lane >> 4;            // which half-wave
    const int batch = blockIdx.x * 16 + m;

    // ---- Preload A1 (permuted W_hh) in WMMA A-layout; A2 as packed dword0 ----
    // A element e <-> K = 16*hi + e   (contiguous order, matches B-layout of h)
    v16h A1[8];
    unsigned int a2w[8];   // dword0 of [W_ih | bias] tile: K=0,1 on lo half-wave
    #pragma unroll
    for (int tau = 0; tau < 8; ++tau) {
        const int g = tau >> 1, p = tau & 1;
        const int grow = 32 * g + 16 * (m >> 3) + 2 * (m & 7) + p;  // permuted gate row
        #pragma unroll
        for (int e = 0; e < 16; ++e) {
            const int k = 16 * hi + e;
            A1[tau][e] = (_Float16)W_hh[grow * HID + k];
        }
        const _Float16 wih = (_Float16)W_ih[grow];
        const _Float16 bb  = (_Float16)(b_ih[grow] + b_hh[grow]);
        a2w[tau] = hi ? 0u : pack_h2(wih, bb);
    }

    // Shared all-zero operand base (8 VGPRs of zeros reused for A2 and B2).
    v16h zv;
    #pragma unroll
    for (int e = 0; e < 16; ++e) zv[e] = (_Float16)0.0f;

    // ---- Recurrent state ----
    v16h  b1 = zv;            // h^T in B-layout (element e <-> K = 16*hi + e)
    float cst[16];            // c state: j = 16*hi + e, batch = lane%16
    #pragma unroll
    for (int e = 0; e < 16; ++e) cst[e] = 0.0f;

    const float* xrow = x + (size_t)batch * Tt;
    float4 xq = *(const float4*)(xrow);      // prefetched chunk of 4 timesteps

    for (int t0 = 0; t0 < Tt; t0 += 4) {
        const int tn = (t0 + 4 < Tt) ? (t0 + 4) : t0;
        const float4 xnext = *(const float4*)(xrow + tn);   // prefetch next chunk
        const float xs4[4] = { xq.x, xq.y, xq.z, xq.w };

        #pragma unroll
        for (int q = 0; q < 4; ++q) {
            // B2 = [x ; 1 ; 0...] : only dword0 (K=0,1) nonzero on lo half-wave
            const unsigned int xw =
                hi ? 0u : pack_h2((_Float16)xs4[q], (_Float16)1.0f);
            const v16h b2 = set_dword0(zv, xw);

            v8f acc[8];
            #pragma unroll
            for (int tau = 0; tau < 8; ++tau) {
                const v16h a2 = set_dword0(zv, a2w[tau]);
                v8f zc = {};
                // d0 = W_ih*x + bias   (rank-1 input/bias contribution)
                v8f d0 = __builtin_amdgcn_wmma_f32_16x16x32_f16(
                    false, a2, false, b2, (short)0, zc, false, false);
                // acc = W_hh*h + d0
                acc[tau] = __builtin_amdgcn_wmma_f32_16x16x32_f16(
                    false, A1[tau], false, b1, (short)0, d0, false, false);
            }

            // Elementwise LSTM cell; all operands lane-local by construction.
            #pragma unroll
            for (int e = 0; e < 16; ++e) {
                const int p = e & 1, r = e >> 1;
                const float iv = fast_sigmoid(acc[0 + p][r]);
                const float fv = fast_sigmoid(acc[2 + p][r]);
                const float gv = fast_tanh   (acc[4 + p][r]);
                const float ov = fast_sigmoid(acc[6 + p][r]);
                const float c  = fv * cst[e] + iv * gv;
                cst[e] = c;
                const float h  = ov * fast_tanh(c);
                b1[e] = (_Float16)h;   // already in B-layout for next step
            }
        }
        xq = xnext;
    }

    // ---- MLP head (once per wave): h_last -> fc1(relu) -> fc2 ----
    __shared__ float hsm[16 * HID];
    #pragma unroll
    for (int e = 0; e < 16; ++e)
        hsm[m * HID + 16 * hi + e] = (float)b1[e];
    __syncthreads();

    if (lane < 16) {
        float accum = fc2_b[0];
        #pragma unroll
        for (int k = 0; k < 16; ++k) {
            float s = fc1_b[k];
            #pragma unroll
            for (int j = 0; j < HID; ++j)
                s += fc1_w[k * HID + j] * hsm[m * HID + j];
            s = s > 0.0f ? s : 0.0f;
            accum += fc2_w[k] * s;
        }
        out[batch] = accum;
    }
}

extern "C" void kernel_launch(void* const* d_in, const int* in_sizes, int n_in,
                              void* d_out, int out_size, void* d_ws, size_t ws_size,
                              hipStream_t stream) {
    const float* x     = (const float*)d_in[0];
    const float* W_ih  = (const float*)d_in[1];
    const float* W_hh  = (const float*)d_in[2];
    const float* b_ih  = (const float*)d_in[3];
    const float* b_hh  = (const float*)d_in[4];
    const float* fc1_w = (const float*)d_in[5];
    const float* fc1_b = (const float*)d_in[6];
    const float* fc2_w = (const float*)d_in[7];
    const float* fc2_b = (const float*)d_in[8];
    float* out = (float*)d_out;

    const int T = T_LEN;
    const int B = in_sizes[0] / T;      // x is [B, T, 1]
    dim3 grid(B / 16), block(32);
    lstm_wmma_kernel<<<grid, block, 0, stream>>>(
        x, W_ih, W_hh, b_ih, b_hh, fc1_w, fc1_b, fc2_w, fc2_b, out, T);
}
